// MultiHeadAttention_74783970558390
// MI455X (gfx1250) — matrix operand order repaired and checked by reading, not compiled, not run on hardware
//
#include <hip/hip_runtime.h>

#define B_  2
#define T_  2048
#define C_  1024
#define H_  16
#define D_  64
#define M_  (B_*T_)   // 4096 rows in the projection GEMMs

typedef __bf16 bf16;
typedef __attribute__((ext_vector_type(16))) __bf16 v16bf;
typedef __attribute__((ext_vector_type(8)))  __bf16 v8bf;
typedef __attribute__((ext_vector_type(4)))  __bf16 v4bf;
typedef __attribute__((ext_vector_type(8)))  float  v8f;

static __device__ __forceinline__ v16bf cat16(v8bf lo, v8bf hi) {
  return __builtin_shufflevector(lo, hi, 0,1,2,3,4,5,6,7,8,9,10,11,12,13,14,15);
}

// A fragment (16x32, row-major src, leading dim lda in elements) at (m0,k0).
// ISA layout: lane r=l&15 -> row m0+r; lanes<16 hold K {0..7,16..23}, lanes>=16 hold {8..15,24..31}.
static __device__ __forceinline__ v16bf load_frag_a(const bf16* A, int lda, int m0, int k0, int lane) {
  int r = lane & 15, g = lane >> 4;
  const bf16* p = A + (size_t)(m0 + r) * lda + (k0 + g * 8);
  v8bf lo = *(const v8bf*)p;          // K = k0+g*8 .. +7
  v8bf hi = *(const v8bf*)(p + 16);   // K = k0+16+g*8 .. +7
  return cat16(lo, hi);
}

// B fragment (32x16) from K-contiguous ("B^T", [N,K] row-major) storage, ldb in elements.
// ISA layout: lane r -> column n0+r; lanes<16 hold K {0..7,16..23}, lanes>=16 hold {8..15,24..31}.
static __device__ __forceinline__ v16bf load_frag_b(const bf16* Bt, int ldb, int n0, int k0, int lane) {
  int r = lane & 15, g = lane >> 4;
  const bf16* p = Bt + (size_t)(n0 + r) * ldb + (k0 + g * 8);
  v8bf lo = *(const v8bf*)p;
  v8bf hi = *(const v8bf*)(p + 16);
  return cat16(lo, hi);
}

static __device__ __forceinline__ v8f wmma_bf16(v16bf a, v16bf b, v8f c) {
  return __builtin_amdgcn_wmma_f32_16x16x32_bf16(false, a, false, b, (short)0, c, false, false);
}

// ---------------- fp32 -> bf16 convert (x) ----------------
__global__ void __launch_bounds__(256) k_cvt(const float* __restrict__ x, bf16* __restrict__ y) {
  size_t i = ((size_t)blockIdx.x * 256 + threadIdx.x) * 4;
  float4 v = *(const float4*)(x + i);
  v4bf o = { (bf16)v.x, (bf16)v.y, (bf16)v.z, (bf16)v.w };
  *(v4bf*)(y + i) = o;
}

// ---------------- weight transpose+convert: Wt[n*C+k] = (bf16)W[k*C+n] ----------------
__global__ void __launch_bounds__(256) k_transpose_cvt(
    const float* __restrict__ w0, const float* __restrict__ w1,
    const float* __restrict__ w2, const float* __restrict__ w3,
    bf16* __restrict__ t0, bf16* __restrict__ t1,
    bf16* __restrict__ t2, bf16* __restrict__ t3) {
  const float* W; bf16* Wt;
  switch (blockIdx.z) {
    case 0:  W = w0; Wt = t0; break;
    case 1:  W = w1; Wt = t1; break;
    case 2:  W = w2; Wt = t2; break;
    default: W = w3; Wt = t3; break;
  }
  __shared__ float tile[32][33];
  int tx = threadIdx.x & 31, ty = threadIdx.x >> 5;   // 32 x 8
  int nb = blockIdx.x * 32, kb = blockIdx.y * 32;
#pragma unroll
  for (int j = 0; j < 4; ++j)
    tile[ty + j * 8][tx] = W[(size_t)(kb + ty + j * 8) * C_ + (nb + tx)];
  __syncthreads();
#pragma unroll
  for (int j = 0; j < 4; ++j)
    Wt[(size_t)(nb + ty + j * 8) * C_ + (kb + tx)] = (bf16)tile[tx][ty + j * 8];
}

// ---------------- fused QKV GEMM: z=0 Q (scaled 1/sqrt(D)), z=1 K, z=2 V (transposed) ----------------
// Workgroup = 8 waves tiling 128x128; each wave owns a 32x64 tile:
// per K-step 2 A-frags + 4 B-frags (12 b128 loads) feed 8 WMMAs (1.5 loads/WMMA).
__global__ void __launch_bounds__(256) k_qkv_gemm(
    const bf16* __restrict__ xb,
    const bf16* __restrict__ WqT, const bf16* __restrict__ WkT, const bf16* __restrict__ WvT,
    const float* __restrict__ bq, const float* __restrict__ bk, const float* __restrict__ bv,
    bf16* __restrict__ Q, bf16* __restrict__ K, bf16* __restrict__ Vt) {
  int z = blockIdx.z;
  const bf16*  Wt   = (z == 0) ? WqT : (z == 1) ? WkT : WvT;
  const float* bias = (z == 0) ? bq  : (z == 1) ? bk  : bv;
  int lane = threadIdx.x & 31, w = threadIdx.x >> 5;
  int m_wave = blockIdx.x * 128 + (w >> 1) * 32;
  int n_wave = blockIdx.y * 128 + (w & 1) * 64;

  v8f acc[2][4] = {};
  for (int kk = 0; kk < C_; kk += 32) {
    v16bf a0 = load_frag_a(xb, C_, m_wave,      kk, lane);
    v16bf a1 = load_frag_a(xb, C_, m_wave + 16, kk, lane);
    v16bf bfr[4];
#pragma unroll
    for (int in = 0; in < 4; ++in)
      bfr[in] = load_frag_b(Wt, C_, n_wave + in * 16, kk, lane);
#pragma unroll
    for (int in = 0; in < 4; ++in) {
      acc[0][in] = wmma_bf16(a0, bfr[in], acc[0][in]);
      acc[1][in] = wmma_bf16(a1, bfr[in], acc[1][in]);
    }
  }

  int r = lane & 15, g = lane >> 4;
  float scale = (z == 0) ? 0.125f : 1.0f;   // 1/sqrt(64) folded into Q
#pragma unroll
  for (int in = 0; in < 4; ++in) {
    int n = n_wave + in * 16 + r;
    float bb = bias[n];
    int h = n >> 6, d = n & (D_ - 1);
#pragma unroll
    for (int im = 0; im < 2; ++im) {
#pragma unroll
      for (int i = 0; i < 8; ++i) {
        int m = m_wave + im * 16 + i + 8 * g;
        int b = m >> 11, t = m & (T_ - 1);
        bf16 o = (bf16)((acc[im][in][i] + bb) * scale);
        if (z == 2)  Vt[(((size_t)b * H_ + h) * D_ + d) * T_ + t] = o;   // [B,H,D,T]
        else if (z == 1) K[(((size_t)b * H_ + h) * T_ + t) * D_ + d] = o; // [B,H,T,D]
        else             Q[(((size_t)b * H_ + h) * T_ + t) * D_ + d] = o;
      }
    }
  }
}

// ---------------- flash attention: one wave per 16-query tile ----------------
__global__ void __launch_bounds__(256) k_attn(
    const bf16* __restrict__ Q, const bf16* __restrict__ K,
    const bf16* __restrict__ Vt, bf16* __restrict__ O) {
  __shared__ bf16 sP[8 * 16 * 32];               // 1KB per wave, per-wave slab (no barriers)
  int lane = threadIdx.x & 31, w = threadIdx.x >> 5;
  int r = lane & 15, g = lane >> 4;
  int bh = blockIdx.y;
  int q0 = blockIdx.x * 128 + w * 16;
  const bf16* Qh = Q  + (size_t)bh * T_ * D_;
  const bf16* Kh = K  + (size_t)bh * T_ * D_;
  const bf16* Vh = Vt + (size_t)bh * D_ * T_;
  bf16* sp = sP + w * (16 * 32);

  v16bf aQ0 = load_frag_a(Qh, D_, q0, 0,  lane);  // d 0..31
  v16bf aQ1 = load_frag_a(Qh, D_, q0, 32, lane);  // d 32..63

  v8f o[4] = {};                                  // 16x64 output accumulator
  float mst[8], lst[8];
#pragma unroll
  for (int i = 0; i < 8; ++i) { mst[i] = -1e30f; lst[i] = 0.f; }

  for (int k0 = 0; k0 < q0 + 16; k0 += 32) {      // causal: only key blocks <= query tile
    v8f s0 = {}, s1 = {};
    s0 = wmma_bf16(aQ0, load_frag_b(Kh, D_, k0,      0,  lane), s0);
    s0 = wmma_bf16(aQ1, load_frag_b(Kh, D_, k0,      32, lane), s0);
    s1 = wmma_bf16(aQ0, load_frag_b(Kh, D_, k0 + 16, 0,  lane), s1);
    s1 = wmma_bf16(aQ1, load_frag_b(Kh, D_, k0 + 16, 32, lane), s1);

#pragma unroll
    for (int i = 0; i < 8; ++i) {
      int row = q0 + i + 8 * g;
      float e0 = (k0 + r      <= row) ? s0[i] : -1e30f;  // causal mask (diag inclusive)
      float e1 = (k0 + 16 + r <= row) ? s1[i] : -1e30f;
      float mx = fmaxf(e0, e1);                           // row max over 16-lane group
      mx = fmaxf(mx, __shfl_xor(mx, 1, 32));
      mx = fmaxf(mx, __shfl_xor(mx, 2, 32));
      mx = fmaxf(mx, __shfl_xor(mx, 4, 32));
      mx = fmaxf(mx, __shfl_xor(mx, 8, 32));
      float mnew  = fmaxf(mst[i], mx);
      float alpha = __expf(mst[i] - mnew);
      float p0 = __expf(e0 - mnew);
      float p1 = __expf(e1 - mnew);
      float rs = p0 + p1;                                 // row sum
      rs += __shfl_xor(rs, 1, 32);
      rs += __shfl_xor(rs, 2, 32);
      rs += __shfl_xor(rs, 4, 32);
      rs += __shfl_xor(rs, 8, 32);
      lst[i] = lst[i] * alpha + rs;
      mst[i] = mnew;
#pragma unroll
      for (int dt = 0; dt < 4; ++dt) o[dt][i] *= alpha;
      sp[(i + 8 * g) * 32 + r]      = (bf16)p0;           // stage P row-major 16x32
      sp[(i + 8 * g) * 32 + 16 + r] = (bf16)p1;
    }

    // reshape P into an A-fragment via LDS (same-wave RAW -> compiler emits s_wait_dscnt)
    const bf16* pp = sp + r * 32 + g * 8;
    v16bf aP = cat16(*(const v8bf*)pp, *(const v8bf*)(pp + 16));
#pragma unroll
    for (int dt = 0; dt < 4; ++dt)
      o[dt] = wmma_bf16(aP, load_frag_b(Vh, T_, dt * 16, k0, lane), o[dt]);
  }

  int b = bh >> 4, h = bh & (H_ - 1);
#pragma unroll
  for (int i = 0; i < 8; ++i) {
    float inv = 1.0f / lst[i];
    int t = q0 + i + 8 * g;
#pragma unroll
    for (int dt = 0; dt < 4; ++dt)
      O[((size_t)b * T_ + t) * C_ + h * D_ + dt * 16 + r] = (bf16)(o[dt][i] * inv);
  }
}

// ---------------- output projection: out = O @ Wo + bo (fp32 out) ----------------
__global__ void __launch_bounds__(256) k_proj_gemm(
    const bf16* __restrict__ Ob, const bf16* __restrict__ WoT,
    const float* __restrict__ bo, float* __restrict__ out) {
  int lane = threadIdx.x & 31, w = threadIdx.x >> 5;
  int m_wave = blockIdx.x * 128 + (w >> 1) * 32;
  int n_wave = blockIdx.y * 128 + (w & 1) * 64;

  v8f acc[2][4] = {};
  for (int kk = 0; kk < C_; kk += 32) {
    v16bf a0 = load_frag_a(Ob, C_, m_wave,      kk, lane);
    v16bf a1 = load_frag_a(Ob, C_, m_wave + 16, kk, lane);
    v16bf bfr[4];
#pragma unroll
    for (int in = 0; in < 4; ++in)
      bfr[in] = load_frag_b(WoT, C_, n_wave + in * 16, kk, lane);
#pragma unroll
    for (int in = 0; in < 4; ++in) {
      acc[0][in] = wmma_bf16(a0, bfr[in], acc[0][in]);
      acc[1][in] = wmma_bf16(a1, bfr[in], acc[1][in]);
    }
  }

  int r = lane & 15, g = lane >> 4;
#pragma unroll
  for (int in = 0; in < 4; ++in) {
    int n = n_wave + in * 16 + r;
    float bb = bo[n];
#pragma unroll
    for (int im = 0; im < 2; ++im) {
#pragma unroll
      for (int i = 0; i < 8; ++i)
        out[(size_t)(m_wave + im * 16 + i + 8 * g) * C_ + n] = acc[im][in][i] + bb;
    }
  }
}

extern "C" void kernel_launch(void* const* d_in, const int* in_sizes, int n_in,
                              void* d_out, int out_size, void* d_ws, size_t ws_size,
                              hipStream_t stream) {
  const float* x  = (const float*)d_in[0];
  const float* Wq = (const float*)d_in[1];
  const float* bq = (const float*)d_in[2];
  const float* Wk = (const float*)d_in[3];
  const float* bk = (const float*)d_in[4];
  const float* Wv = (const float*)d_in[5];
  const float* bv = (const float*)d_in[6];
  const float* Wo = (const float*)d_in[7];
  const float* bo = (const float*)d_in[8];
  float* out = (float*)d_out;

  char* ws = (char*)d_ws;
  size_t off = 0;
  auto wsalloc = [&](size_t bytes) {
    char* p = ws + off;
    off += (bytes + 255) & ~(size_t)255;
    return p;
  };
  bf16* xb  = (bf16*)wsalloc((size_t)M_ * C_ * 2);          // 8 MB
  bf16* WqT = (bf16*)wsalloc((size_t)C_ * C_ * 2);          // 2 MB x4
  bf16* WkT = (bf16*)wsalloc((size_t)C_ * C_ * 2);
  bf16* WvT = (bf16*)wsalloc((size_t)C_ * C_ * 2);
  bf16* WoT = (bf16*)wsalloc((size_t)C_ * C_ * 2);
  bf16* Qb  = (bf16*)wsalloc((size_t)B_ * H_ * T_ * D_ * 2); // 8 MB each
  bf16* Kb  = (bf16*)wsalloc((size_t)B_ * H_ * T_ * D_ * 2);
  bf16* Vtb = (bf16*)wsalloc((size_t)B_ * H_ * T_ * D_ * 2);
  bf16* Ob  = (bf16*)wsalloc((size_t)M_ * C_ * 2);           // 8 MB

  // 1. x -> bf16
  k_cvt<<<dim3((M_ * C_) / (256 * 4)), 256, 0, stream>>>(x, xb);
  // 2. weights -> bf16, transposed to [N,K]
  k_transpose_cvt<<<dim3(C_ / 32, C_ / 32, 4), 256, 0, stream>>>(
      Wq, Wk, Wv, Wo, WqT, WkT, WvT, WoT);
  // 3. Q,K,V projections (Q pre-scaled by 1/sqrt(D); V stored transposed)
  k_qkv_gemm<<<dim3(M_ / 128, C_ / 128, 3), 256, 0, stream>>>(
      xb, WqT, WkT, WvT, bq, bk, bv, Qb, Kb, Vtb);
  // 4. causal flash attention
  k_attn<<<dim3(T_ / 128, B_ * H_), 256, 0, stream>>>(Qb, Kb, Vtb, Ob);
  // 5. output projection -> fp32 d_out
  k_proj_gemm<<<dim3(M_ / 128, C_ / 128), 256, 0, stream>>>(Ob, WoT, bo, out);
}
